// MultiHeadAttention_47021301956741
// MI455X (gfx1250) — matrix operand order repaired and checked by reading, not compiled, not run on hardware
//
#include <hip/hip_runtime.h>

// ---------------------------------------------------------------------------
// CDNA5 (gfx1250, wave32) MHA: bf16 WMMA GEMMs + flash attention.
// - v_wmma_f32_16x16x32_bf16 for all matrix math (f32 accumulate)
// - double-buffered LDS tiles everywhere
// - GLOBAL_LOAD_ASYNC_TO_LDS_B128 (ASYNCcnt) for Q/K staging in attention
// ---------------------------------------------------------------------------

typedef __bf16 bf16_t;
typedef bf16_t v16bf __attribute__((ext_vector_type(16)));
typedef float  v8f   __attribute__((ext_vector_type(8)));
typedef unsigned int   uint2v  __attribute__((ext_vector_type(2)));
typedef unsigned int   uint4v  __attribute__((ext_vector_type(4)));
typedef float          float4v __attribute__((ext_vector_type(4)));

union Frag {            // one WMMA A/B operand: 16 bf16 = 8 VGPRs
    v16bf  v;
    uint4v q[2];
};

__device__ __forceinline__ unsigned short f32_to_bf16(float f) {
    unsigned int u = __float_as_uint(f);
    unsigned int r = 0x7FFFu + ((u >> 16) & 1u);   // round-to-nearest-even
    return (unsigned short)((u + r) >> 16);
}
__device__ __forceinline__ unsigned pack_bf16x2(float lo, float hi) {
    return ((unsigned)f32_to_bf16(hi) << 16) | f32_to_bf16(lo);
}

// Low 32 bits of a flat pointer to __shared__ == LDS byte offset.
__device__ __forceinline__ unsigned lds_off32(const void* p) {
    return (unsigned)(unsigned long long)p;
}
// Async global->LDS copy, 16B per lane, tracked on ASYNCcnt.
__device__ __forceinline__ void async_copy_b128(unsigned ldsoff, const void* gaddr) {
    asm volatile("global_load_async_to_lds_b128 %0, %1, off"
                 :: "v"(ldsoff), "v"(gaddr) : "memory");
}
__device__ __forceinline__ void wait_asynccnt0() {
    asm volatile("s_wait_asynccnt 0" ::: "memory");
}

// ---------------------------------------------------------------------------
// GEMM with fused f32->bf16 staging:  out[M,N] = A[M,K] * Bw[K,N] + bias[N]
// Block: 256 threads (8 waves), tile 128x128, k-step 32, double-buffered LDS.
// Wave grid 4x2: each wave computes 32 rows x 64 cols = 2x4 WMMA tiles.
// ---------------------------------------------------------------------------
#define BM 128
#define BN 128
#define BKK 32

template <bool OUT_BF16>
__global__ __launch_bounds__(256)
void wmma_gemm_bias(const float* __restrict__ A, const float* __restrict__ Bw,
                    const float* __restrict__ bias, void* __restrict__ out,
                    int M, int N, int K)
{
    __shared__ unsigned short As[2][BM][BKK];   // [m][k] bf16
    __shared__ unsigned short Bs[2][BN][BKK];   // [n][k] bf16 (transposed)

    const int t      = threadIdx.x;
    const int wave   = t >> 5;
    const int lane   = t & 31;
    const int lanelo = lane & 15;
    const int half   = lane >> 4;

    const int m0 = blockIdx.y * BM;
    const int n0 = blockIdx.x * BN;
    const int wm = wave >> 1;    // 0..3
    const int wn = wave & 1;     // 0..1

    v8f acc[2][4];
#pragma unroll
    for (int i = 0; i < 2; ++i)
#pragma unroll
        for (int j = 0; j < 4; ++j) acc[i][j] = {};

    float4v aR[4], bR[4];

    auto gload = [&](int k0) {
#pragma unroll
        for (int i = 0; i < 4; ++i) {              // A tile: 128x32 f32
            int lin = t + i * 256;
            int row = lin >> 3;                    // 8 float4 per 32-wide row
            int c4  = lin & 7;
            aR[i] = *reinterpret_cast<const float4v*>(
                &A[(size_t)(m0 + row) * K + k0 + c4 * 4]);
        }
#pragma unroll
        for (int i = 0; i < 4; ++i) {              // B tile: 32x128 f32
            int lin  = t + i * 256;
            int krow = lin >> 5;                   // 32 float4 per 128-wide row
            int nc   = (lin & 31) * 4;
            bR[i] = *reinterpret_cast<const float4v*>(
                &Bw[(size_t)(k0 + krow) * N + n0 + nc]);
        }
    };
    auto stage = [&](int buf) {
#pragma unroll
        for (int i = 0; i < 4; ++i) {
            int lin = t + i * 256;
            int row = lin >> 3;
            int c4  = lin & 7;
            uint2v p;
            p.x = pack_bf16x2(aR[i].x, aR[i].y);
            p.y = pack_bf16x2(aR[i].z, aR[i].w);
            *reinterpret_cast<uint2v*>(&As[buf][row][c4 * 4]) = p;
        }
#pragma unroll
        for (int i = 0; i < 4; ++i) {              // transpose into Bs[n][k]
            int lin  = t + i * 256;
            int krow = lin >> 5;
            int nc   = (lin & 31) * 4;
            Bs[buf][nc + 0][krow] = f32_to_bf16(bR[i].x);
            Bs[buf][nc + 1][krow] = f32_to_bf16(bR[i].y);
            Bs[buf][nc + 2][krow] = f32_to_bf16(bR[i].z);
            Bs[buf][nc + 3][krow] = f32_to_bf16(bR[i].w);
        }
    };

    gload(0);
    stage(0);
    __syncthreads();

    int cur = 0;
    for (int k0 = 0; k0 < K; k0 += BKK) {
        const int  nxt  = cur ^ 1;
        const bool more = (k0 + BKK) < K;
        if (more) gload(k0 + BKK);                 // overlap with WMMA below

        Frag a[2], b[4];
#pragma unroll
        for (int i = 0; i < 2; ++i) {
            int row = wm * 32 + i * 16 + lanelo;
            a[i].q[0] = *reinterpret_cast<const uint4v*>(&As[cur][row][half * 8]);
            a[i].q[1] = *reinterpret_cast<const uint4v*>(&As[cur][row][16 + half * 8]);
        }
#pragma unroll
        for (int j = 0; j < 4; ++j) {
            int col = wn * 64 + j * 16 + lanelo;
            b[j].q[0] = *reinterpret_cast<const uint4v*>(&Bs[cur][col][half * 8]);
            b[j].q[1] = *reinterpret_cast<const uint4v*>(&Bs[cur][col][16 + half * 8]);
        }
#pragma unroll
        for (int i = 0; i < 2; ++i)
#pragma unroll
            for (int j = 0; j < 4; ++j)
                acc[i][j] = __builtin_amdgcn_wmma_f32_16x16x32_bf16(
                    false, a[i].v, false, b[j].v, (short)0, acc[i][j], false, false);

        if (more) stage(nxt);
        __syncthreads();
        cur = nxt;
    }

    // ---- epilogue: bias + store (bf16 or f32)
#pragma unroll
    for (int i = 0; i < 2; ++i) {
#pragma unroll
        for (int j = 0; j < 4; ++j) {
            int   col = n0 + wn * 64 + j * 16 + lanelo;
            float bv  = bias[col];
#pragma unroll
            for (int r = 0; r < 8; ++r) {
                int   row = m0 + wm * 32 + i * 16 + r + 8 * half;
                float val = acc[i][j][r] + bv;
                if constexpr (OUT_BF16)
                    reinterpret_cast<unsigned short*>(out)[(size_t)row * N + col] =
                        f32_to_bf16(val);
                else
                    reinterpret_cast<float*>(out)[(size_t)row * N + col] = val;
            }
        }
    }
}

// ---------------------------------------------------------------------------
// Flash attention: block = (b,h) + 128 query rows; 8 waves x 16 rows.
// Q/K staged via async global->LDS (ASYNCcnt); K/V double-buffered.
// ---------------------------------------------------------------------------
#define S_LEN 2048
#define HD 64
#define QT 128
#define KT 64

__global__ __launch_bounds__(256)
void wmma_flash_attn(const unsigned short* __restrict__ Qp,
                     const unsigned short* __restrict__ Kp,
                     const unsigned short* __restrict__ Vp,
                     float* __restrict__ ctx)
{
    __shared__ unsigned short Qs[QT][HD];           // [q][d]
    __shared__ unsigned short Ks[2][KT][HD];        // [kpos][d], ping-pong
    __shared__ unsigned short Vt[2][HD][KT];        // [d][kpos], ping-pong
    __shared__ unsigned short Ps[8][16][KT];        // per-wave P slab

    const int t      = threadIdx.x;
    const int wave   = t >> 5;
    const int lane   = t & 31;
    const int lanelo = lane & 15;
    const int half   = lane >> 4;

    const int bh = blockIdx.x;           // 0..31
    const int b  = bh >> 4;
    const int h  = bh & 15;
    const int q0 = blockIdx.y * QT;

    const size_t rowbase = (size_t)b * S_LEN;
    const int    colbase = h * HD;

    auto stageK_async = [&](int buf, int kc) {
#pragma unroll
        for (int i = 0; i < 2; ++i) {
            int lin = t + i * 256;
            int row = lin >> 3;
            int c8  = (lin & 7) * 8;
            async_copy_b128(lds_off32(&Ks[buf][row][c8]),
                            &Kp[(rowbase + kc + row) * 1024 + colbase + c8]);
        }
    };
    uint4v vR[2];
    auto loadV = [&](int kc) {
#pragma unroll
        for (int i = 0; i < 2; ++i) {
            int lin = t + i * 256;
            int p   = lin >> 3;
            int d8  = (lin & 7) * 8;
            vR[i] = *reinterpret_cast<const uint4v*>(
                &Vp[(rowbase + kc + p) * 1024 + colbase + d8]);
        }
    };
    auto storeVt = [&](int buf) {
#pragma unroll
        for (int i = 0; i < 2; ++i) {
            int lin = t + i * 256;
            int p   = lin >> 3;
            int d8  = (lin & 7) * 8;
            const unsigned short* e = reinterpret_cast<const unsigned short*>(&vR[i]);
#pragma unroll
            for (int j = 0; j < 8; ++j) Vt[buf][d8 + j][p] = e[j];
        }
    };

    // ---- prologue: Q tile + chunk 0, all staged asynchronously where possible
#pragma unroll
    for (int i = 0; i < 4; ++i) {
        int lin = t + i * 256;
        int row = lin >> 3;
        int c8  = (lin & 7) * 8;
        async_copy_b128(lds_off32(&Qs[row][c8]),
                        &Qp[(rowbase + q0 + row) * 1024 + colbase + c8]);
    }
    stageK_async(0, 0);
    loadV(0);
    storeVt(0);
    wait_asynccnt0();
    __syncthreads();

    v8f acc_o[4];
#pragma unroll
    for (int j = 0; j < 4; ++j) acc_o[j] = {};
    float m_run[8], l_run[8];
#pragma unroll
    for (int r = 0; r < 8; ++r) { m_run[r] = -__builtin_inff(); l_run[r] = 0.0f; }

    int cur = 0;
    for (int kc = 0; kc < S_LEN; kc += KT) {
        const int  nxt  = cur ^ 1;
        const bool more = (kc + KT) < S_LEN;
        if (more) {                       // prefetch next chunk during compute
            stageK_async(nxt, kc + KT);
            loadV(kc + KT);
        }

        // ---- scores: S = Q_tile(wave) x K^T  (4 subtiles of 16 key positions)
        v8f sc[4];
#pragma unroll
        for (int j = 0; j < 4; ++j) sc[j] = {};
#pragma unroll
        for (int kb = 0; kb < 2; ++kb) {
            Frag aq;
            int qrow = wave * 16 + lanelo;
            aq.q[0] = *reinterpret_cast<const uint4v*>(&Qs[qrow][kb * 32 + half * 8]);
            aq.q[1] = *reinterpret_cast<const uint4v*>(&Qs[qrow][kb * 32 + 16 + half * 8]);
#pragma unroll
            for (int j = 0; j < 4; ++j) {
                Frag bk;
                int kp = j * 16 + lanelo;
                bk.q[0] = *reinterpret_cast<const uint4v*>(&Ks[cur][kp][kb * 32 + half * 8]);
                bk.q[1] = *reinterpret_cast<const uint4v*>(&Ks[cur][kp][kb * 32 + 16 + half * 8]);
                sc[j] = __builtin_amdgcn_wmma_f32_16x16x32_bf16(
                    false, aq.v, false, bk.v, (short)0, sc[j], false, false);
            }
        }

        // ---- online softmax (row = vgpr-index + 8*laneHalf; 16-lane row groups)
#pragma unroll
        for (int r = 0; r < 8; ++r) {
            float mx = -__builtin_inff();
#pragma unroll
            for (int j = 0; j < 4; ++j) { sc[j][r] *= 0.125f; mx = fmaxf(mx, sc[j][r]); }
            mx = fmaxf(mx, __shfl_xor(mx, 1, 32));
            mx = fmaxf(mx, __shfl_xor(mx, 2, 32));
            mx = fmaxf(mx, __shfl_xor(mx, 4, 32));
            mx = fmaxf(mx, __shfl_xor(mx, 8, 32));
            float m_new = fmaxf(m_run[r], mx);
            float alpha = __expf(m_run[r] - m_new);
            float rs = 0.0f;
#pragma unroll
            for (int j = 0; j < 4; ++j) {
                float p = __expf(sc[j][r] - m_new);
                sc[j][r] = p;
                rs += p;
            }
            rs += __shfl_xor(rs, 1, 32);
            rs += __shfl_xor(rs, 2, 32);
            rs += __shfl_xor(rs, 4, 32);
            rs += __shfl_xor(rs, 8, 32);
            l_run[r] = l_run[r] * alpha + rs;
            m_run[r] = m_new;
#pragma unroll
            for (int j = 0; j < 4; ++j) acc_o[j][r] *= alpha;
        }

        // ---- P: C-layout regs -> bf16 A-layout via per-wave LDS slab
#pragma unroll
        for (int j = 0; j < 4; ++j)
#pragma unroll
            for (int r = 0; r < 8; ++r)
                Ps[wave][r + 8 * half][j * 16 + lanelo] = f32_to_bf16(sc[j][r]);

        // ---- ctx += P x V
#pragma unroll
        for (int kb = 0; kb < 2; ++kb) {
            Frag ap;
            ap.q[0] = *reinterpret_cast<const uint4v*>(&Ps[wave][lanelo][kb * 32 + half * 8]);
            ap.q[1] = *reinterpret_cast<const uint4v*>(&Ps[wave][lanelo][kb * 32 + 16 + half * 8]);
#pragma unroll
            for (int j = 0; j < 4; ++j) {
                Frag bv;
                int d = j * 16 + lanelo;
                bv.q[0] = *reinterpret_cast<const uint4v*>(&Vt[cur][d][kb * 32 + half * 8]);
                bv.q[1] = *reinterpret_cast<const uint4v*>(&Vt[cur][d][kb * 32 + 16 + half * 8]);
                acc_o[j] = __builtin_amdgcn_wmma_f32_16x16x32_bf16(
                    false, ap.v, false, bv.v, (short)0, acc_o[j], false, false);
            }
        }

        if (more) storeVt(nxt);
        wait_asynccnt0();
        __syncthreads();
        cur = nxt;
    }

    // ---- epilogue: normalize and write f32 ctx [B*S, D]
#pragma unroll
    for (int j = 0; j < 4; ++j) {
#pragma unroll
        for (int r = 0; r < 8; ++r) {
            int   qrow = q0 + wave * 16 + r + 8 * half;
            float val  = acc_o[j][r] / l_run[r];
            ctx[(rowbase + qrow) * 1024 + colbase + j * 16 + lanelo] = val;
        }
    }
}

// ---------------------------------------------------------------------------
// Launch: 3 projection GEMMs (bf16 out) -> flash attention -> output GEMM.
// ---------------------------------------------------------------------------
extern "C" void kernel_launch(void* const* d_in, const int* in_sizes, int n_in,
                              void* d_out, int out_size, void* d_ws, size_t ws_size,
                              hipStream_t stream)
{
    const float* q  = (const float*)d_in[0];
    const float* k  = (const float*)d_in[1];
    const float* v  = (const float*)d_in[2];
    const float* Wq = (const float*)d_in[3];
    const float* bq = (const float*)d_in[4];
    const float* Wk = (const float*)d_in[5];
    const float* bk = (const float*)d_in[6];
    const float* Wv = (const float*)d_in[7];
    const float* bv = (const float*)d_in[8];
    const float* Wo = (const float*)d_in[9];
    const float* bo = (const float*)d_in[10];

    const int M = 4096, N = 1024, K = 1024;   // B*S = 4096, D_MODEL = 1024

    char* ws = (char*)d_ws;
    unsigned short* Qp = (unsigned short*)(ws);                       //  8 MB bf16
    unsigned short* Kp = (unsigned short*)(ws + (size_t)M * N * 2);   //  8 MB bf16
    unsigned short* Vp = (unsigned short*)(ws + (size_t)M * N * 4);   //  8 MB bf16
    float*          ct = (float*)        (ws + (size_t)M * N * 6);    // 16 MB f32

    dim3 blk(256);
    dim3 gproj(N / BN, M / BM);      // (8, 32)
    dim3 gattn(32, S_LEN / QT);      // (B*H, 16)

    wmma_gemm_bias<true ><<<gproj, blk, 0, stream>>>(q,  Wq, bq, Qp,    M, N, K);
    wmma_gemm_bias<true ><<<gproj, blk, 0, stream>>>(k,  Wk, bk, Kp,    M, N, K);
    wmma_gemm_bias<true ><<<gproj, blk, 0, stream>>>(v,  Wv, bv, Vp,    M, N, K);
    wmma_flash_attn      <<<gattn, blk, 0, stream>>>(Qp, Kp, Vp, ct);
    wmma_gemm_bias<false><<<gproj, blk, 0, stream>>>(ct, Wo, bo, d_out, M, N, K);
}
